// Mamba_80376017977908
// MI455X (gfx1250) — matrix operand order repaired and checked by reading, not compiled, not run on hardware
//
#include <hip/hip_runtime.h>
#include <hip/hip_bf16.h>

// ---------------------------------------------------------------------------
// Mamba block forward for MI455X (gfx1250, wave32).
// GEMMs run on v_wmma_f32_16x16x32_f16 (f16 in, f32 accumulate).
// ---------------------------------------------------------------------------

#define D_MODEL  1024
#define D_STATE  16
#define D_CONV   4
#define D_INNER  2048
#define BATCH    4
#define SEQ      2048
#define NROWS    (BATCH * SEQ)          // 8192 rows of activations

typedef __attribute__((ext_vector_type(16))) _Float16 v16h;
typedef __attribute__((ext_vector_type(8)))  _Float16 v8h;
typedef __attribute__((ext_vector_type(8)))  float    v8f;

// ---------------------------------------------------------------------------
// fp32 -> f16 conversion
// ---------------------------------------------------------------------------
__global__ __launch_bounds__(256) void cvt_f32_to_f16(
    const float* __restrict__ src, _Float16* __restrict__ dst, int n) {
  int i = blockIdx.x * 256 + threadIdx.x;
  if (i < n) dst[i] = (_Float16)src[i];
}

// ---------------------------------------------------------------------------
// WMMA GEMM:  C[M,N] (f32) = A[M,K] (f16, row-major) @ W[N,K]^T (f16, row-major)
// Wave computes a (16*MT)(M) x (16*NT)(N) strip; B fragments are reused across
// the MT row-tiles (1.5 b128 loads per WMMA at MT=2,NT=4).
// Block = 4 waves stacked in M -> block tile = 64*MT rows x 64*NT/4... grid:
//   grid = (N/(16*NT), M/(64*MT)), block = 128.
// Fragment layouts per CDNA5 ISA 7.12.2:
//   A (16x32 f16): lane L holds row M=L&15; halves 0..7 -> K = k0+(L>>4)*8+i,
//                  halves 8..15 -> K = k0+(L>>4)*8+16+i.
//   B (32x16 f16): lane L holds col N=L&15; same K mapping as A.
//   C (16x16 f32): reg v, lane L -> (M = v + 8*(L>>4), N = L&15).
// All loads are 16B/32B contiguous runs -> global_load_b128.
// ---------------------------------------------------------------------------
template <int MT, int NT>
__global__ __launch_bounds__(128) void gemm_wmma_f16(
    const _Float16* __restrict__ A, const _Float16* __restrict__ W,
    float* __restrict__ C, int M, int N, int K) {
  const int lane = threadIdx.x & 31;
  const int wave = threadIdx.x >> 5;
  const int half = lane >> 4;   // 0: lanes 0-15, 1: lanes 16-31
  const int l16  = lane & 15;

  const int m0 = (blockIdx.y * 4 + wave) * (16 * MT);
  const int n0 = blockIdx.x * (NT * 16);

  v8f acc[MT][NT];
#pragma unroll
  for (int mt = 0; mt < MT; ++mt)
#pragma unroll
    for (int t = 0; t < NT; ++t)
#pragma unroll
      for (int i = 0; i < 8; ++i) acc[mt][t][i] = 0.0f;

  const _Float16* __restrict__ Arow[MT];
#pragma unroll
  for (int mt = 0; mt < MT; ++mt)
    Arow[mt] = A + (size_t)(m0 + mt * 16 + l16) * K;

  for (int k0 = 0; k0 < K; k0 += 32) {
    union { v16h v; v8h h[2]; } af[MT];
#pragma unroll
    for (int mt = 0; mt < MT; ++mt) {
      af[mt].h[0] = *(const v8h*)(Arow[mt] + k0 + half * 8);       // K base..+7
      af[mt].h[1] = *(const v8h*)(Arow[mt] + k0 + half * 8 + 16);  // +16..+23
    }
#pragma unroll
    for (int t = 0; t < NT; ++t) {
      const _Float16* __restrict__ Wp =
          W + (size_t)(n0 + t * 16 + l16) * K + k0 + half * 8;
      union { v16h v; v8h h[2]; } bf;
      bf.h[0] = *(const v8h*)(Wp);
      bf.h[1] = *(const v8h*)(Wp + 16);
#pragma unroll
      for (int mt = 0; mt < MT; ++mt)
        acc[mt][t] = __builtin_amdgcn_wmma_f32_16x16x32_f16(
            false, af[mt].v, false, bf.v, (short)0, acc[mt][t], false, false);
    }
  }

#pragma unroll
  for (int mt = 0; mt < MT; ++mt)
#pragma unroll
    for (int t = 0; t < NT; ++t) {
      float* __restrict__ Cp =
          C + (size_t)(m0 + mt * 16 + half * 8) * N + n0 + t * 16 + l16;
#pragma unroll
      for (int v = 0; v < 8; ++v) Cp[(size_t)v * N] = acc[mt][t][v];
    }
}

// ---------------------------------------------------------------------------
// Causal depthwise conv (taps=4) + bias + SiLU.  Reads xi half of xz,
// writes u as f16 (feeds GEMM2 and scan).
// ---------------------------------------------------------------------------
__global__ __launch_bounds__(256) void conv_silu_kernel(
    const float* __restrict__ xz, const float* __restrict__ cw,
    const float* __restrict__ cb, _Float16* __restrict__ ub) {
  int idx = blockIdx.x * 256 + threadIdx.x;          // (row, e), e fastest
  if (idx >= NROWS * D_INNER) return;
  const int e   = idx & (D_INNER - 1);
  const int row = idx >> 11;                          // /2048
  const int l   = row & (SEQ - 1);
  const int b   = row >> 11;

  float acc = cb[e];
  const float w0 = cw[e * D_CONV + 0];
  const float w1 = cw[e * D_CONV + 1];
  const float w2 = cw[e * D_CONV + 2];
  const float w3 = cw[e * D_CONV + 3];
#define XI(ls) xz[((size_t)(b * SEQ + (ls)) << 12) + e]   // stride 4096
  if (l >= 3) acc += w0 * XI(l - 3);
  if (l >= 2) acc += w1 * XI(l - 2);
  if (l >= 1) acc += w2 * XI(l - 1);
  acc += w3 * XI(l);
#undef XI
  const float u = acc / (1.0f + __expf(-acc));        // SiLU
  ub[idx] = (_Float16)u;
}

// ---------------------------------------------------------------------------
// SSM parameter prep: from bcd[row][48] = [B(16) | C(16) | dt_pre(16)]
// delta = softplus(dt_pre + b_dt);  dA = exp(delta * -exp(A_log));  dB = delta*B
// ---------------------------------------------------------------------------
__global__ __launch_bounds__(256) void ssm_prep_kernel(
    const float* __restrict__ bcd, const float* __restrict__ b_dt,
    const float* __restrict__ A_log,
    float* __restrict__ dAo, float* __restrict__ dBo, float* __restrict__ Co) {
  int idx = blockIdx.x * 256 + threadIdx.x;           // (row, n)
  if (idx >= NROWS * D_STATE) return;
  const int n   = idx & (D_STATE - 1);
  const int row = idx >> 4;

  const float bv = bcd[row * 48 + n];
  const float cv = bcd[row * 48 + 16 + n];
  float dt = bcd[row * 48 + 32 + n] + b_dt[n];
  dt = (dt > 20.0f) ? dt : log1pf(__expf(dt));        // softplus
  const float Ac = -__expf(A_log[n]);
  dAo[idx] = __expf(dt * Ac);
  dBo[idx] = dt * bv;
  Co[idx]  = cv;
}

// ---------------------------------------------------------------------------
// Selective scan, fused with skip (+u*D), gate (*silu(z)), f16 convert.
// One thread per (b, e) channel; h[16] lives in registers; per-step
// dA/dB/C (shared across e) come in as three float4x4 broadcast loads.
// grid = (D_INNER/256, BATCH), block = 256.
// ---------------------------------------------------------------------------
__global__ __launch_bounds__(256) void scan_gate_kernel(
    const float* __restrict__ dA, const float* __restrict__ dB,
    const float* __restrict__ Cm, const _Float16* __restrict__ ub,
    const float* __restrict__ xz, const float* __restrict__ Dv,
    _Float16* __restrict__ gated) {
  const int b = blockIdx.y;
  const int e = blockIdx.x * 256 + threadIdx.x;
  const float De = Dv[e];

  float h[D_STATE];
#pragma unroll
  for (int n = 0; n < D_STATE; ++n) h[n] = 0.0f;

  for (int l = 0; l < SEQ; ++l) {
    const int row = b * SEQ + l;
    const float4* __restrict__ pA = (const float4*)(dA + row * 16);
    const float4* __restrict__ pB = (const float4*)(dB + row * 16);
    const float4* __restrict__ pC = (const float4*)(Cm + row * 16);
    const float u = (float)ub[((size_t)row << 11) + e];

    float y = 0.0f;
#pragma unroll
    for (int q = 0; q < 4; ++q) {
      const float4 a4 = pA[q], b4 = pB[q], c4 = pC[q];
      h[4 * q + 0] = a4.x * h[4 * q + 0] + u * b4.x;  y += h[4 * q + 0] * c4.x;
      h[4 * q + 1] = a4.y * h[4 * q + 1] + u * b4.y;  y += h[4 * q + 1] * c4.y;
      h[4 * q + 2] = a4.z * h[4 * q + 2] + u * b4.z;  y += h[4 * q + 2] * c4.z;
      h[4 * q + 3] = a4.w * h[4 * q + 3] + u * b4.w;  y += h[4 * q + 3] * c4.w;
    }

    const float z = xz[((size_t)row << 12) + D_INNER + e];
    const float g = z / (1.0f + __expf(-z));          // silu(z)
    gated[((size_t)row << 11) + e] = (_Float16)((y + u * De) * g);
  }
}

// ---------------------------------------------------------------------------
// Host-side orchestration
// ---------------------------------------------------------------------------
static inline size_t align256(size_t x) { return (x + 255) & ~(size_t)255; }

extern "C" void kernel_launch(void* const* d_in, const int* in_sizes, int n_in,
                              void* d_out, int out_size, void* d_ws, size_t ws_size,
                              hipStream_t stream) {
  (void)in_sizes; (void)n_in; (void)out_size; (void)ws_size;

  const float* x      = (const float*)d_in[0];   // (B,L,1024)
  const float* W_in   = (const float*)d_in[1];   // (4096,1024)
  const float* conv_w = (const float*)d_in[2];   // (2048,1,4)
  const float* conv_b = (const float*)d_in[3];   // (2048,)
  const float* W_x    = (const float*)d_in[4];   // (32,2048)
  const float* W_dt   = (const float*)d_in[5];   // (16,2048)
  const float* b_dt   = (const float*)d_in[6];   // (16,)
  const float* A_log  = (const float*)d_in[7];   // (16,)
  const float* Dv     = (const float*)d_in[8];   // (2048,)
  const float* W_out  = (const float*)d_in[9];   // (1024,2048)
  float* out = (float*)d_out;                    // (B,L,1024) f32

  // ---- workspace carve ----
  char* p = (char*)d_ws;
  size_t off = 0;
  auto carve = [&](size_t bytes) { void* r = p + off; off += align256(bytes); return r; };

  _Float16* xb     = (_Float16*)carve((size_t)NROWS * D_MODEL * 2);       // 16.8 MB
  _Float16* Wb_in  = (_Float16*)carve((size_t)2 * D_INNER * D_MODEL * 2); //  8.4 MB
  float*    xz     = (float*)   carve((size_t)NROWS * 2 * D_INNER * 4);   //  134 MB
  _Float16* ub     = (_Float16*)carve((size_t)NROWS * D_INNER * 2);       // 33.6 MB
  _Float16* Wcat   = (_Float16*)carve((size_t)48 * D_INNER * 2);          //  0.2 MB
  float*    bcd    = (float*)   carve((size_t)NROWS * 48 * 4);            //  1.6 MB
  float*    dA     = (float*)   carve((size_t)NROWS * D_STATE * 4);
  float*    dB     = (float*)   carve((size_t)NROWS * D_STATE * 4);
  float*    Cc     = (float*)   carve((size_t)NROWS * D_STATE * 4);
  _Float16* gated  = (_Float16*)carve((size_t)NROWS * D_INNER * 2);       // 33.6 MB
  _Float16* Wb_out = (_Float16*)carve((size_t)D_MODEL * D_INNER * 2);     //  4.2 MB

  auto cvt = [&](const float* s, _Float16* d, int n) {
    cvt_f32_to_f16<<<dim3((n + 255) / 256), dim3(256), 0, stream>>>(s, d, n);
  };

  // ---- 0. fp32 -> f16 staging ----
  cvt(x,     xb,     NROWS * D_MODEL);
  cvt(W_in,  Wb_in,  2 * D_INNER * D_MODEL);
  cvt(W_x,   Wcat,            32 * D_INNER);     // rows 0..31 : B,C proj
  cvt(W_dt,  Wcat + 32 * D_INNER, 16 * D_INNER); // rows 32..47 : dt proj
  cvt(W_out, Wb_out, D_MODEL * D_INNER);

  // ---- 1. in_proj: xz = x @ W_in^T   (8192 x 4096, K=1024) ----
  // wave tile 32x64; block tile 128(M) x 64(N)
  gemm_wmma_f16<2, 4><<<dim3((2 * D_INNER) / 64, NROWS / 128), dim3(128), 0, stream>>>(
      xb, Wb_in, xz, NROWS, 2 * D_INNER, D_MODEL);

  // ---- 2. depthwise causal conv + SiLU -> u (f16) ----
  conv_silu_kernel<<<dim3((NROWS * D_INNER) / 256), dim3(256), 0, stream>>>(
      xz, conv_w, conv_b, ub);

  // ---- 3. bcd = u @ [W_x;W_dt]^T   (8192 x 48, K=2048) ----
  gemm_wmma_f16<1, 1><<<dim3(3, NROWS / 64), dim3(128), 0, stream>>>(
      ub, Wcat, bcd, NROWS, 48, D_INNER);

  // ---- 4. softplus / exp -> dA, dB, C ----
  ssm_prep_kernel<<<dim3((NROWS * D_STATE) / 256), dim3(256), 0, stream>>>(
      bcd, b_dt, A_log, dA, dB, Cc);

  // ---- 5. selective scan + skip + gate -> gated (f16) ----
  scan_gate_kernel<<<dim3(D_INNER / 256, BATCH), dim3(256), 0, stream>>>(
      dA, dB, Cc, ub, xz, Dv, gated);

  // ---- 6. out = gated @ W_out^T   (8192 x 1024, K=2048) ----
  gemm_wmma_f16<2, 4><<<dim3(D_MODEL / 64, NROWS / 128), dim3(128), 0, stream>>>(
      gated, Wb_out, out, NROWS, D_MODEL, D_INNER);
}
